// GCNII_29841432772821
// MI455X (gfx1250) — matrix operand order repaired and checked by reading, not compiled, not run on hardware
//
#include <hip/hip_runtime.h>
#include <hip/hip_bf16.h>
#include <math.h>

// ---------------- problem constants (match reference) ----------------
static constexpr int NN   = 50000;
static constexpr int EE   = 1600000;
static constexpr int DIN  = 512;
static constexpr int HID  = 128;
static constexpr int DOUT = 64;
static constexpr int LAY  = 8;

// ---------------- WMMA vector types ----------------
typedef __bf16 v16bf __attribute__((ext_vector_type(16)));
typedef float  v8f   __attribute__((ext_vector_type(8)));

union V16 {
    v16bf    v;
    unsigned p[8];          // packed bf16 pairs (element 2i in low half)
};

// pack two f32 -> one dword of two bf16 (round-to-nearest, ties away).
// v_perm_b32 selects bytes {b.3,b.2,a.3,a.2} -> {bf16(hi), bf16(lo)}.
__device__ __forceinline__ unsigned pack_bf16(float lo, float hi) {
    unsigned a = __float_as_uint(lo) + 0x8000u;
    unsigned b = __float_as_uint(hi) + 0x8000u;
    return __builtin_amdgcn_perm(b, a, 0x07060302u);
}

// ======================================================================
// Kernel 1: H0 = H = relu(feature @ W0 + b0)     [N,512] x [512,128]
// Block: 256 thr = 8 waves; each wave -> 16 rows x 128 cols.
// W0 staged to LDS in 128-K chunks, pre-swizzled into WMMA B lane layout.
// ======================================================================
__global__ void __launch_bounds__(256)
gcnii_gemm_in(const float* __restrict__ feat, const float* __restrict__ W0,
              const float* __restrict__ b0, float* __restrict__ H0,
              float* __restrict__ H, int n)
{
    __shared__ __align__(32) unsigned sB[4 * 8 * 32 * 8]; // 32 KB, 8 dw/slot

    const int tid  = threadIdx.x;
    const int wave = tid >> 5;
    const int lane = tid & 31;
    const int half = (lane < 16) ? 0 : 1;
    const int mbase = blockIdx.x * 128 + wave * 16;

    v8f acc[8] = {};

    int arow = mbase + (lane & 15);
    if (arow >= n) arow = n - 1;                 // clamp reads, guard writes
    const float* aptr = feat + (size_t)arow * DIN;

    for (int kc = 0; kc < 4; ++kc) {             // 4 chunks of K=128
        // ---- stage + swizzle W0[kc*128 .. +128][0..128] ----
        for (int slot = tid; slot < 1024; slot += 256) {
            const int sl = slot & 31;
            const int nt = (slot >> 5) & 7;
            const int kb = slot >> 8;
            const int ncol  = nt * 16 + (sl & 15);
            const int kbase = kc * 128 + kb * 32 + ((sl < 16) ? 0 : 8);
            unsigned* p = sB + slot * 8;
#pragma unroll
            for (int i = 0; i < 8; ++i)
                p[i] = pack_bf16(W0[(size_t)(kbase + 2 * i + ((i & 4) << 1))     * HID + ncol],
                                 W0[(size_t)(kbase + 2 * i + ((i & 4) << 1) + 1) * HID + ncol]);
        }
        __syncthreads();

#pragma unroll
        for (int kb = 0; kb < 4; ++kb) {
            const int abase = kc * 128 + kb * 32 + half * 8;
            float4 a0 = *(const float4*)(aptr + abase);
            float4 a1 = *(const float4*)(aptr + abase + 4);
            float4 a2 = *(const float4*)(aptr + abase + 16);
            float4 a3 = *(const float4*)(aptr + abase + 20);
            V16 av;
            av.p[0] = pack_bf16(a0.x, a0.y);
            av.p[1] = pack_bf16(a0.z, a0.w);
            av.p[2] = pack_bf16(a1.x, a1.y);
            av.p[3] = pack_bf16(a1.z, a1.w);
            av.p[4] = pack_bf16(a2.x, a2.y);
            av.p[5] = pack_bf16(a2.z, a2.w);
            av.p[6] = pack_bf16(a3.x, a3.y);
            av.p[7] = pack_bf16(a3.z, a3.w);
#pragma unroll
            for (int nt = 0; nt < 8; ++nt) {
                v16bf bv = *(const v16bf*)(sB + ((kb * 8 + nt) * 32 + lane) * 8);
                acc[nt] = __builtin_amdgcn_wmma_f32_16x16x32_bf16(
                    false, av.v, false, bv, (short)0, acc[nt], false, false);
            }
        }
        __syncthreads();
    }

    // ---- epilogue: bias + relu, write H0 and H ----
    const int nc    = lane & 15;
    const int rbase = mbase + half * 8;
#pragma unroll
    for (int t = 0; t < 8; ++t) {
        const int ncol = t * 16 + nc;
        const float bias = b0[ncol];
#pragma unroll
        for (int r = 0; r < 8; ++r) {
            const int m = rbase + r;
            if (m < n) {
                float v = acc[t][r] + bias;
                v = v > 0.f ? v : 0.f;
                H0[(size_t)m * HID + ncol] = v;
                H [(size_t)m * HID + ncol] = v;
            }
        }
    }
}

// ======================================================================
// Kernel 2: zero Hm
// ======================================================================
__global__ void gcnii_zero(float4* __restrict__ p, int n4)
{
    int i = blockIdx.x * blockDim.x + threadIdx.x;
    if (i < n4) p[i] = make_float4(0.f, 0.f, 0.f, 0.f);
}

// ======================================================================
// Kernel 3: SpMM  Hm[dst] += vals * H[src].
// Each wave owns 32 edges: lanes load edge metadata coalesced, then a
// 32-step loop broadcasts (src,dst,val) via shfl while all lanes gather
// float4 of H[src] and issue 4 L2-resident f32 atomics.
// ======================================================================
__global__ void __launch_bounds__(256)
gcnii_spmm(const int* __restrict__ src, const int* __restrict__ dst,
           const float* __restrict__ vals, const float* __restrict__ H,
           float* __restrict__ Hm, int e)
{
    const int gwave = (blockIdx.x * blockDim.x + threadIdx.x) >> 5;
    const int lane  = threadIdx.x & 31;
    const int base  = gwave * 32;
    if (base >= e) return;

    const int myEdge = base + lane;
    const bool valid = myEdge < e;
    const int   ms = valid ? src[myEdge]  : 0;
    const int   md = valid ? dst[myEdge]  : 0;
    const float mv = valid ? vals[myEdge] : 0.f;

    const int cnt = (e - base < 32) ? (e - base) : 32;
    for (int i = 0; i < cnt; ++i) {
        const int   s = __shfl(ms, i, 32);
        const int   d = __shfl(md, i, 32);
        const float v = __shfl(mv, i, 32);
        const float4 h = *(const float4*)(H + (size_t)s * HID + lane * 4);
        float* o = Hm + (size_t)d * HID + lane * 4;
        atomicAdd(o + 0, v * h.x);
        atomicAdd(o + 1, v * h.y);
        atomicAdd(o + 2, v * h.z);
        atomicAdd(o + 3, v * h.w);
    }
}

// ======================================================================
// Kernel 4: fused layer update
//   sup = 0.9*Hm + 0.1*H0 ;  H = relu((1-b)*sup + b*(sup @ Wc))
// Wc (128x128) fully staged/swizzled in 32KB LDS.
// ======================================================================
__global__ void __launch_bounds__(256)
gcnii_layer(const float* __restrict__ Hm, const float* __restrict__ H0,
            const float* __restrict__ W, float beta,
            float* __restrict__ H, int n)
{
    __shared__ __align__(32) unsigned sB[4 * 8 * 32 * 8]; // 32 KB

    const int tid  = threadIdx.x;
    const int wave = tid >> 5;
    const int lane = tid & 31;
    const int half = (lane < 16) ? 0 : 1;
    const int mbase = blockIdx.x * 128 + wave * 16;

    for (int slot = tid; slot < 1024; slot += 256) {
        const int sl = slot & 31;
        const int nt = (slot >> 5) & 7;
        const int kb = slot >> 8;
        const int ncol  = nt * 16 + (sl & 15);
        const int kbase = kb * 32 + ((sl < 16) ? 0 : 8);
        unsigned* p = sB + slot * 8;
#pragma unroll
        for (int i = 0; i < 8; ++i)
            p[i] = pack_bf16(W[(size_t)(kbase + 2 * i + ((i & 4) << 1))     * HID + ncol],
                             W[(size_t)(kbase + 2 * i + ((i & 4) << 1) + 1) * HID + ncol]);
    }
    __syncthreads();

    v8f acc[8] = {};

    int arow = mbase + (lane & 15);
    if (arow >= n) arow = n - 1;
    const float* mp = Hm + (size_t)arow * HID;
    const float* zp = H0 + (size_t)arow * HID;

#pragma unroll
    for (int kb = 0; kb < 4; ++kb) {
        const int abase = kb * 32 + half * 8;
        float4 m0 = *(const float4*)(mp + abase);
        float4 m1 = *(const float4*)(mp + abase + 4);
        float4 m2 = *(const float4*)(mp + abase + 16);
        float4 m3 = *(const float4*)(mp + abase + 20);
        float4 z0 = *(const float4*)(zp + abase);
        float4 z1 = *(const float4*)(zp + abase + 4);
        float4 z2 = *(const float4*)(zp + abase + 16);
        float4 z3 = *(const float4*)(zp + abase + 20);
        V16 av;
        av.p[0] = pack_bf16(0.9f*m0.x + 0.1f*z0.x, 0.9f*m0.y + 0.1f*z0.y);
        av.p[1] = pack_bf16(0.9f*m0.z + 0.1f*z0.z, 0.9f*m0.w + 0.1f*z0.w);
        av.p[2] = pack_bf16(0.9f*m1.x + 0.1f*z1.x, 0.9f*m1.y + 0.1f*z1.y);
        av.p[3] = pack_bf16(0.9f*m1.z + 0.1f*z1.z, 0.9f*m1.w + 0.1f*z1.w);
        av.p[4] = pack_bf16(0.9f*m2.x + 0.1f*z2.x, 0.9f*m2.y + 0.1f*z2.y);
        av.p[5] = pack_bf16(0.9f*m2.z + 0.1f*z2.z, 0.9f*m2.w + 0.1f*z2.w);
        av.p[6] = pack_bf16(0.9f*m3.x + 0.1f*z3.x, 0.9f*m3.y + 0.1f*z3.y);
        av.p[7] = pack_bf16(0.9f*m3.z + 0.1f*z3.z, 0.9f*m3.w + 0.1f*z3.w);
#pragma unroll
        for (int nt = 0; nt < 8; ++nt) {
            v16bf bv = *(const v16bf*)(sB + ((kb * 8 + nt) * 32 + lane) * 8);
            acc[nt] = __builtin_amdgcn_wmma_f32_16x16x32_bf16(
                false, av.v, false, bv, (short)0, acc[nt], false, false);
        }
    }

    const int nc    = lane & 15;
    const int rbase = mbase + half * 8;
    const float omb = 1.f - beta;
#pragma unroll
    for (int t = 0; t < 8; ++t) {
        const int ncol = t * 16 + nc;
#pragma unroll
        for (int r = 0; r < 8; ++r) {
            const int m = rbase + r;
            if (m < n) {
                const size_t idx = (size_t)m * HID + ncol;
                float sup = 0.9f * Hm[idx] + 0.1f * H0[idx];
                float v = omb * sup + beta * acc[t][r];
                v = v > 0.f ? v : 0.f;
                H[idx] = v;
            }
        }
    }
}

// ======================================================================
// Kernel 5: logits = H @ W1 + b1 ; out = log_softmax(logits)
// Wave: 16 rows x 64 cols. Row reductions via __shfl_xor within 16-lane
// halves (masks 1..8 never cross the half boundary).
// ======================================================================
__global__ void __launch_bounds__(256)
gcnii_out(const float* __restrict__ H, const float* __restrict__ W1,
          const float* __restrict__ b1, float* __restrict__ out, int n)
{
    __shared__ __align__(32) unsigned sB[4 * 4 * 32 * 8]; // 16 KB

    const int tid  = threadIdx.x;
    const int wave = tid >> 5;
    const int lane = tid & 31;
    const int half = (lane < 16) ? 0 : 1;
    const int mbase = blockIdx.x * 128 + wave * 16;

    for (int slot = tid; slot < 512; slot += 256) {
        const int sl = slot & 31;
        const int nt = (slot >> 5) & 3;
        const int kb = slot >> 7;
        const int ncol  = nt * 16 + (sl & 15);
        const int kbase = kb * 32 + ((sl < 16) ? 0 : 8);
        unsigned* p = sB + slot * 8;
#pragma unroll
        for (int i = 0; i < 8; ++i)
            p[i] = pack_bf16(W1[(size_t)(kbase + 2 * i + ((i & 4) << 1))     * DOUT + ncol],
                             W1[(size_t)(kbase + 2 * i + ((i & 4) << 1) + 1) * DOUT + ncol]);
    }
    __syncthreads();

    v8f acc[4] = {};

    int arow = mbase + (lane & 15);
    if (arow >= n) arow = n - 1;
    const float* aptr = H + (size_t)arow * HID;

#pragma unroll
    for (int kb = 0; kb < 4; ++kb) {
        const int abase = kb * 32 + half * 8;
        float4 a0 = *(const float4*)(aptr + abase);
        float4 a1 = *(const float4*)(aptr + abase + 4);
        float4 a2 = *(const float4*)(aptr + abase + 16);
        float4 a3 = *(const float4*)(aptr + abase + 20);
        V16 av;
        av.p[0] = pack_bf16(a0.x, a0.y);
        av.p[1] = pack_bf16(a0.z, a0.w);
        av.p[2] = pack_bf16(a1.x, a1.y);
        av.p[3] = pack_bf16(a1.z, a1.w);
        av.p[4] = pack_bf16(a2.x, a2.y);
        av.p[5] = pack_bf16(a2.z, a2.w);
        av.p[6] = pack_bf16(a3.x, a3.y);
        av.p[7] = pack_bf16(a3.z, a3.w);
#pragma unroll
        for (int nt = 0; nt < 4; ++nt) {
            v16bf bv = *(const v16bf*)(sB + ((kb * 4 + nt) * 32 + lane) * 8);
            acc[nt] = __builtin_amdgcn_wmma_f32_16x16x32_bf16(
                false, av.v, false, bv, (short)0, acc[nt], false, false);
        }
    }

    const int nc    = lane & 15;
    const int rbase = mbase + half * 8;
    const float bb0 = b1[nc], bb1 = b1[16 + nc], bb2 = b1[32 + nc], bb3 = b1[48 + nc];
#pragma unroll
    for (int r = 0; r < 8; ++r) {
        float l0 = acc[0][r] + bb0;
        float l1 = acc[1][r] + bb1;
        float l2 = acc[2][r] + bb2;
        float l3 = acc[3][r] + bb3;
        float vmax = fmaxf(fmaxf(l0, l1), fmaxf(l2, l3));
#pragma unroll
        for (int off = 8; off >= 1; off >>= 1)
            vmax = fmaxf(vmax, __shfl_xor(vmax, off, 32));
        float s = __expf(l0 - vmax) + __expf(l1 - vmax) +
                  __expf(l2 - vmax) + __expf(l3 - vmax);
#pragma unroll
        for (int off = 8; off >= 1; off >>= 1)
            s += __shfl_xor(s, off, 32);
        const float lse = vmax + __logf(s);
        const int m = rbase + r;
        if (m < n) {
            float* o = out + (size_t)m * DOUT;
            o[nc]      = l0 - lse;
            o[16 + nc] = l1 - lse;
            o[32 + nc] = l2 - lse;
            o[48 + nc] = l3 - lse;
        }
    }
}

// ======================================================================
extern "C" void kernel_launch(void* const* d_in, const int* in_sizes, int n_in,
                              void* d_out, int out_size, void* d_ws, size_t ws_size,
                              hipStream_t stream)
{
    const float* feature = (const float*)d_in[0];
    const int*   src     = (const int*)  d_in[1];
    const int*   dst     = (const int*)  d_in[2];
    const float* vals    = (const float*)d_in[3];
    const float* W0      = (const float*)d_in[4];
    const float* b0      = (const float*)d_in[5];
    const float* Wc      = (const float*)d_in[6];
    const float* W1      = (const float*)d_in[7];
    const float* b1      = (const float*)d_in[8];
    float* out = (float*)d_out;

    // workspace: H0 | H | Hm  (3 x 25.6 MB f32)
    float* H0 = (float*)d_ws;
    float* H  = H0 + (size_t)NN * HID;
    float* Hm = H  + (size_t)NN * HID;

    const int gemmBlocks = (NN + 127) / 128;
    const int n4         = NN * HID / 4;
    const int zeroBlocks = (n4 + 255) / 256;
    const int spmmWaves  = (EE + 31) / 32;
    const int spmmBlocks = (spmmWaves * 32 + 255) / 256;

    gcnii_gemm_in<<<gemmBlocks, 256, 0, stream>>>(feature, W0, b0, H0, H, NN);

    for (int l = 0; l < LAY; ++l) {
        const float beta = logf(0.5f / (float)(l + 1) + 1.0f);
        gcnii_zero<<<zeroBlocks, 256, 0, stream>>>((float4*)Hm, n4);
        gcnii_spmm<<<spmmBlocks, 256, 0, stream>>>(src, dst, vals, H, Hm, EE);
        gcnii_layer<<<gemmBlocks, 256, 0, stream>>>(
            Hm, H0, Wc + (size_t)l * HID * HID, beta, H, NN);
    }

    gcnii_out<<<gemmBlocks, 256, 0, stream>>>(H, W1, b1, out, NN);
}
